// DeepSeekAttention_1228360646729
// MI455X (gfx1250) — compile-verified, matrix operand order checked by reading, not run on hardware
//
#include <hip/hip_runtime.h>

// ---------------------------------------------------------------------------
// DeepSeek-style attention for MI455X (gfx1250, wave32, WMMA f32_16x16x32_f16)
//   B=2, S=2048, D=1024, H=16, HD=64
// Latent transforms folded into projection weights; flash-attention keeps the
// 537MB score tensor in LDS/registers; all GEMMs use v_wmma_f32_16x16x32_f16.
// Tile staging uses the Tensor Data Mover (TENSOR_LOAD_TO_LDS + TENSORcnt)
// when the toolchain exposes it, with LDS row padding done by the TDM itself.
// ---------------------------------------------------------------------------

typedef __attribute__((ext_vector_type(16))) _Float16 v16h;
typedef __attribute__((ext_vector_type(8)))  float    v8f;

#define DM    1024
#define NH    16
#define HD    64
#define SEQ   2048
#define BATCH 2

#if defined(__AMDGCN__) && __has_builtin(__builtin_amdgcn_tensor_load_to_lds) \
                        && __has_builtin(__builtin_amdgcn_s_wait_tensorcnt)
#define USE_TDM 1
#else
#define USE_TDM 0
#endif

// ---- WMMA wrapper ---------------------------------------------------------
__device__ __forceinline__ v8f wmma_f16(v16h a, v16h b, v8f c) {
  // 8 args: (neg_a, A, neg_b, B, c_mod, C, reuse_a, reuse_b)
  return __builtin_amdgcn_wmma_f32_16x16x32_f16(false, a, false, b,
                                                (short)0, c, false, false);
}

#if USE_TDM
typedef __attribute__((ext_vector_type(4))) unsigned int u32x4;
typedef __attribute__((ext_vector_type(4))) int          i32x4;
typedef __attribute__((ext_vector_type(8))) int          i32x8;

// 2-D TDM load: tile_h rows x tile_w elements (2B each) from a row-major
// tensor with row stride `stride_elems`, into LDS at byte offset lds_addr,
// with `pad_amt_code`+1 DWORDs of LDS padding every 2^(pad_int_code+1) DWORDs.
// Tensor dims are set equal to tile dims (no OOB clipping needed here).
__device__ __forceinline__ void tdm_load_2d(const void* gptr,
                                            unsigned lds_addr,
                                            unsigned tile_w, unsigned tile_h,
                                            unsigned stride_elems,
                                            unsigned pad_int_code,
                                            unsigned pad_amt_code) {
  unsigned long long ga = (unsigned long long)(uintptr_t)gptr;
  u32x4 g0;
  g0.x = 1u;                                       // count=1, user descriptor
  g0.y = lds_addr;                                 // lds_addr (bytes)
  g0.z = (unsigned)(ga & 0xFFFFFFFFu);             // global_addr[31:0]
  g0.w = (unsigned)((ga >> 32) & 0x01FFFFFFu)      // global_addr[56:32]
         | 0x80000000u;                            // type=2 ("image")
  i32x8 g1;
  g1[0] = (int)((1u << 16)                         // data_size = 2 bytes
                | (1u << 20)                       // pad_enable
                | (pad_int_code << 22)             // pad_interval
                | (pad_amt_code << 25));           // pad_amount
  g1[1] = (int)((tile_w & 0xFFFFu) << 16);         // tensor_dim0[15:0]
  g1[2] = (int)(((tile_w >> 16) & 0xFFFFu)         // tensor_dim0[31:16]
                | ((tile_h & 0xFFFFu) << 16));     // tensor_dim1[15:0]
  g1[3] = (int)(((tile_h >> 16) & 0xFFFFu)         // tensor_dim1[31:16]
                | ((tile_w & 0xFFFFu) << 16));     // tile_dim0
  g1[4] = (int)(tile_h & 0xFFFFu);                 // tile_dim1 (tile_dim2=0)
  g1[5] = (int)stride_elems;                       // tensor_dim0_stride[31:0]
  g1[6] = 0;                                       // stride[47:32], dim1_stride
  g1[7] = 0;
  i32x4 gz = {0, 0, 0, 0};
#if __clang_major__ >= 23
  i32x8 gz8 = {0, 0, 0, 0, 0, 0, 0, 0};
  __builtin_amdgcn_tensor_load_to_lds(g0, g1, gz, gz, gz8, 0);
#else
  __builtin_amdgcn_tensor_load_to_lds(g0, g1, gz, gz, 0);
#endif
}
#endif  // USE_TDM

// ---- Fragment loader ------------------------------------------------------
// 16-bit A (16x32, MxK) fragment per ISA layout:
//   lanes 0-15 : row M=lane,    K = {k0..k0+7} U {k0+16..k0+23}
//   lanes 16-31: row M=lane-16, K = {k0+8..k0+15} U {k0+24..k0+31}
// B (32x16, KxN) fragments use the identical pattern with "row" = column N.
// Requires ld % 8 == 0 and k0 % 8 == 0 for 16-byte aligned loads.
__device__ __forceinline__ v16h load_frag(const _Float16* __restrict__ p,
                                          int ld, int row0, int k0) {
  const int lane = threadIdx.x & 31;
  const _Float16* q = p + (size_t)(row0 + (lane & 15)) * ld
                        + k0 + ((lane & 16) ? 8 : 0);
  union { v16h f; uint4 u[2]; } U;
  U.u[0] = *(const uint4*)(q);
  U.u[1] = *(const uint4*)(q + 16);
  return U.f;
}

// ---- Small prep kernels ---------------------------------------------------
__global__ void cvt_f32_f16(const float* __restrict__ in,
                            _Float16* __restrict__ out, int n) {
  int i = blockIdx.x * 256 + threadIdx.x;
  if (i < n) out[i] = (_Float16)in[i];
}

// out[h*64+i, c] = sum_j Wl[i,j] * W[h*64+j, c]   (fold latent into projection)
__global__ void fold_latent(const float* __restrict__ Wl,
                            const float* __restrict__ W,
                            _Float16* __restrict__ out) {
  int idx = blockIdx.x * 256 + threadIdx.x;   // 1024*1024 threads
  int n = idx >> 10, c = idx & 1023;
  int hb = n >> 6,  i = n & 63;
  const float* wl = Wl + (size_t)i * 64;
  const float* wr = W + (size_t)hb * 64 * DM + c;
  float acc = 0.f;
#pragma unroll 8
  for (int j = 0; j < 64; ++j) acc += wl[j] * wr[(size_t)j * DM];
  out[(size_t)n * DM + c] = (_Float16)acc;
}

// ---- 128x128-tile WMMA GEMM:  C[M,N] = A[M,K] @ W[N,K]^T (+bias[col%64]) ---
// Double-buffered LDS; tiles staged by the TDM (wave 0 issues, TENSORcnt
// pipelined against WMMA compute) or by per-lane b128 copies as fallback.
template <bool HAS_BIAS, bool OUT_F16>
__global__ __launch_bounds__(256) void gemm128(
    const _Float16* __restrict__ A, const _Float16* __restrict__ W,
    const float* __restrict__ bias, void* __restrict__ Cout,
    int K, int N) {
  __shared__ _Float16 lA[2][128 * 40];  // 128 x 32 (+8 pad) per buffer
  __shared__ _Float16 lB[2][128 * 40];

  const int tid = threadIdx.x;
  const int m0 = blockIdx.x * 128;
  const int n0 = blockIdx.y * 128;
  const int w = tid >> 5, lane = tid & 31;
  const int wm = (w & 3) * 32;        // 4 waves along M (32 rows each)
  const int wn = (w >> 2) * 64;       // 2 waves along N (64 cols each)

  v8f zz = {};
  v8f acc[2][4];
#pragma unroll
  for (int i = 0; i < 2; ++i)
#pragma unroll
    for (int j = 0; j < 4; ++j) acc[i][j] = zz;

  const int ksteps = K >> 5;

#if USE_TDM
  if (w == 0) {   // prologue: stage first K-slab (pad 4 DWORDs / 16 DWORDs)
    tdm_load_2d(&A[(size_t)m0 * K], (unsigned)(uintptr_t)&lA[0][0],
                32u, 128u, (unsigned)K, 3u, 3u);
    tdm_load_2d(&W[(size_t)n0 * K], (unsigned)(uintptr_t)&lB[0][0],
                32u, 128u, (unsigned)K, 3u, 3u);
  }
#endif

  for (int ks = 0; ks < ksteps; ++ks) {
    const int buf = ks & 1;
#if USE_TDM
    if (w == 0) __builtin_amdgcn_s_wait_tensorcnt(0);
    __syncthreads();                  // data visible; prev-buf readers done
    if (w == 0 && ks + 1 < ksteps) {  // stage next slab during compute
      tdm_load_2d(&A[(size_t)m0 * K + (ks + 1) * 32],
                  (unsigned)(uintptr_t)&lA[buf ^ 1][0],
                  32u, 128u, (unsigned)K, 3u, 3u);
      tdm_load_2d(&W[(size_t)n0 * K + (ks + 1) * 32],
                  (unsigned)(uintptr_t)&lB[buf ^ 1][0],
                  32u, 128u, (unsigned)K, 3u, 3u);
    }
#else
    __syncthreads();
#pragma unroll
    for (int t = 0; t < 2; ++t) {     // 512 16B vectors per tile, 2/thread
      int vi = tid + t * 256;
      int row = vi >> 2, c8 = (vi & 3) * 8;
      *(uint4*)&lA[buf][row * 40 + c8] =
          *(const uint4*)&A[(size_t)(m0 + row) * K + ks * 32 + c8];
      *(uint4*)&lB[buf][row * 40 + c8] =
          *(const uint4*)&W[(size_t)(n0 + row) * K + ks * 32 + c8];
    }
    if (ks + 1 < ksteps) {
      __builtin_prefetch(&A[(size_t)(m0 + (tid >> 2)) * K + (ks + 1) * 32 + (tid & 3) * 8], 0, 0);
      __builtin_prefetch(&W[(size_t)(n0 + (tid >> 2)) * K + (ks + 1) * 32 + (tid & 3) * 8], 0, 0);
    }
    __syncthreads();
#endif

    v16h af[2], bf[4];
#pragma unroll
    for (int i = 0; i < 2; ++i) af[i] = load_frag(lA[buf], 40, wm + i * 16, 0);
#pragma unroll
    for (int j = 0; j < 4; ++j) bf[j] = load_frag(lB[buf], 40, wn + j * 16, 0);
#pragma unroll
    for (int i = 0; i < 2; ++i)
#pragma unroll
      for (int j = 0; j < 4; ++j)
        acc[i][j] = wmma_f16(af[i], bf[j], acc[i][j]);
  }

  // C/D layout: lane(0-15)=col, VGPR r = row r (lanes<16) or row r+8 (>=16)
  const int hi = (lane & 16) ? 8 : 0;
  const int cn = lane & 15;
#pragma unroll
  for (int i = 0; i < 2; ++i)
#pragma unroll
    for (int j = 0; j < 4; ++j)
#pragma unroll
      for (int r = 0; r < 8; ++r) {
        int row = m0 + wm + i * 16 + hi + r;
        int col = n0 + wn + j * 16 + cn;
        float v = acc[i][j][r];
        if (HAS_BIAS) v += bias[col & 63];
        if (OUT_F16)
          ((_Float16*)Cout)[(size_t)row * N + col] = (_Float16)v;
        else
          ((float*)Cout)[(size_t)row * N + col] = v;
      }
}

// ---- Flash attention: one block = 128 queries of one (b,h) ----------------
__global__ __launch_bounds__(256) void flash_attn(
    const _Float16* __restrict__ lq, const _Float16* __restrict__ lk,
    const _Float16* __restrict__ vv, _Float16* __restrict__ att) {
  __shared__ _Float16 ldsK[128 * 72];    // key tile,   t-major (t x d)
  __shared__ _Float16 ldsVt[64 * 136];   // value tile, d-major (d x t)
  __shared__ _Float16 ldsP[128 * 136];   // probs (f16) for re-fragmenting

  const int tid = threadIdx.x, w = tid >> 5, lane = tid & 31;
  const int qt = blockIdx.x & 15;
  const int h  = (blockIdx.x >> 4) & 15;
  const int b  = blockIdx.x >> 8;
  const int q0 = qt * 128;
  const int hi = (lane & 16) ? 8 : 0;
  const int cn = lane & 15;

  // Per-wave 16 query rows; lq A-fragments loaded once, reused for all keys.
  const _Float16* lqbase =
      lq + (size_t)(b * SEQ + q0 + w * 16) * DM + h * HD;
  v16h aq[2];
  aq[0] = load_frag(lqbase, DM, 0, 0);
  aq[1] = load_frag(lqbase, DM, 0, 32);

  float mrow[8], lrow[8];
#pragma unroll
  for (int r = 0; r < 8; ++r) { mrow[r] = -1e30f; lrow[r] = 0.f; }
  v8f zz = {};
  v8f facc[4];
#pragma unroll
  for (int j = 0; j < 4; ++j) facc[j] = zz;

  for (int kt = 0; kt < 16; ++kt) {
    const int t0 = kt * 128;
    __syncthreads();                   // all waves done reading prev tiles
#if USE_TDM
    // TDM stages the K tile (128 x 64 halves, stride DM, pad 4 DWORDs every
    // 32 DWORDs -> LDS stride 72 halves) while all threads transpose V.
    if (w == 0)
      tdm_load_2d(&lk[(size_t)(b * SEQ + t0) * DM + h * HD],
                  (unsigned)(uintptr_t)&ldsK[0], 64u, 128u, (unsigned)DM,
                  4u, 3u);
#pragma unroll
    for (int t = 0; t < 4; ++t) {
      int vi = tid + t * 256;
      int trow = vi >> 3, d8 = (vi & 7) * 8;
      union { uint4 u; _Float16 hl[8]; } U;
      U.u = *(const uint4*)&vv[(size_t)(b * SEQ + t0 + trow) * DM + h * HD + d8];
#pragma unroll
      for (int qd = 0; qd < 8; ++qd)
        ldsVt[(d8 + qd) * 136 + trow] = U.hl[qd];
    }
    if (w == 0) __builtin_amdgcn_s_wait_tensorcnt(0);
#else
#pragma unroll
    for (int t = 0; t < 4; ++t) {
      int vi = tid + t * 256;          // 1024 16B vectors
      int trow = vi >> 3, d8 = (vi & 7) * 8;
      const size_t g = (size_t)(b * SEQ + t0 + trow) * DM + h * HD + d8;
      *(uint4*)&ldsK[trow * 72 + d8] = *(const uint4*)&lk[g];
      union { uint4 u; _Float16 hl[8]; } U;
      U.u = *(const uint4*)&vv[g];
#pragma unroll
      for (int qd = 0; qd < 8; ++qd)
        ldsVt[(d8 + qd) * 136 + trow] = U.hl[qd];
    }
#endif
    __syncthreads();

    // scores: 16 query rows x 128 keys, K=HD=64 (2 WMMA per 16x16 tile)
    v8f sc[8];
#pragma unroll
    for (int jt = 0; jt < 8; ++jt) {
      v8f s = zz;
      s = wmma_f16(aq[0], load_frag(ldsK, 72, jt * 16, 0), s);
      s = wmma_f16(aq[1], load_frag(ldsK, 72, jt * 16, 32), s);
      sc[jt] = s;
    }

    // online softmax (scale 1/sqrt(64)=0.125); row reductions across the
    // 16-lane N-group via shfl_xor masks 1,2,4,8 (stay within the group)
    float scal[8];
#pragma unroll
    for (int r = 0; r < 8; ++r) {
      float mx = -1e30f;
#pragma unroll
      for (int jt = 0; jt < 8; ++jt) mx = fmaxf(mx, sc[jt][r]);
      mx *= 0.125f;
#pragma unroll
      for (int off = 1; off <= 8; off <<= 1)
        mx = fmaxf(mx, __shfl_xor(mx, off, 32));
      float mn = fmaxf(mrow[r], mx);
      float s0 = __expf(mrow[r] - mn);
      float rs = 0.f;
#pragma unroll
      for (int jt = 0; jt < 8; ++jt) {
        float p = __expf(sc[jt][r] * 0.125f - mn);
        sc[jt][r] = p;
        rs += p;
      }
#pragma unroll
      for (int off = 1; off <= 8; off <<= 1) rs += __shfl_xor(rs, off, 32);
      lrow[r] = lrow[r] * s0 + rs;
      mrow[r] = mn;
      scal[r] = s0;
    }
#pragma unroll
    for (int j = 0; j < 4; ++j)
#pragma unroll
      for (int r = 0; r < 8; ++r) facc[j][r] *= scal[r];

    // spill P (f16) to this wave's private LDS rows, re-fragment as A matrix
#pragma unroll
    for (int jt = 0; jt < 8; ++jt)
#pragma unroll
      for (int r = 0; r < 8; ++r)
        ldsP[(w * 16 + hi + r) * 136 + jt * 16 + cn] = (_Float16)sc[jt][r];

    // out += P(16x128) @ V(128x64): 4 K-chunks x 4 d-tiles
#pragma unroll
    for (int c = 0; c < 4; ++c) {
      v16h pf = load_frag(ldsP, 136, w * 16, c * 32);
#pragma unroll
      for (int j = 0; j < 4; ++j)
        facc[j] = wmma_f16(pf, load_frag(ldsVt, 136, j * 16, c * 32), facc[j]);
    }
  }

  // normalize and emit attended (B,S,H,HD) in f16
#pragma unroll
  for (int r = 0; r < 8; ++r) {
    const float inv = 1.f / lrow[r];
    const size_t row = (size_t)(b * SEQ + q0 + w * 16 + hi + r) * DM + h * HD;
#pragma unroll
    for (int j = 0; j < 4; ++j)
      att[row + j * 16 + cn] = (_Float16)(facc[j][r] * inv);
  }
}

// ---------------------------------------------------------------------------
extern "C" void kernel_launch(void* const* d_in, const int* in_sizes, int n_in,
                              void* d_out, int out_size, void* d_ws,
                              size_t ws_size, hipStream_t stream) {
  (void)in_sizes; (void)n_in; (void)out_size; (void)ws_size;
  const float* x   = (const float*)d_in[0];
  const float* Wq  = (const float*)d_in[1];
  const float* Wk  = (const float*)d_in[2];
  const float* Wv  = (const float*)d_in[3];
  const float* Wo  = (const float*)d_in[4];
  const float* Wlq = (const float*)d_in[5];
  const float* blq = (const float*)d_in[6];
  const float* Wlk = (const float*)d_in[7];
  const float* blk = (const float*)d_in[8];

  char* ws = (char*)d_ws;
  const size_t MB = 1u << 20;
  _Float16* xh  = (_Float16*)(ws + 0 * MB);   // x in f16          (8 MB)
  _Float16* Wqe = (_Float16*)(ws + 8 * MB);   // folded Wlq*Wq     (2 MB)
  _Float16* Wke = (_Float16*)(ws + 10 * MB);  // folded Wlk*Wk     (2 MB)
  _Float16* Wvh = (_Float16*)(ws + 12 * MB);  // Wv f16            (2 MB)
  _Float16* Woh = (_Float16*)(ws + 14 * MB);  // Wo f16            (2 MB)
  _Float16* lqb = (_Float16*)(ws + 16 * MB);  // latent Q          (8 MB)
  _Float16* lkb = (_Float16*)(ws + 24 * MB);  // latent K          (8 MB)
  _Float16* vb  = (_Float16*)(ws + 32 * MB);  // V                 (8 MB)
  _Float16* ab  = (_Float16*)(ws + 40 * MB);  // attended          (8 MB)

  const int nx = BATCH * SEQ * DM;            // 4M elements
  cvt_f32_f16<<<nx / 256, 256, 0, stream>>>(x, xh, nx);
  fold_latent<<<(DM * DM) / 256, 256, 0, stream>>>(Wlq, Wq, Wqe);
  fold_latent<<<(DM * DM) / 256, 256, 0, stream>>>(Wlk, Wk, Wke);
  cvt_f32_f16<<<(DM * DM) / 256, 256, 0, stream>>>(Wv, Wvh, DM * DM);
  cvt_f32_f16<<<(DM * DM) / 256, 256, 0, stream>>>(Wo, Woh, DM * DM);

  dim3 g((BATCH * SEQ) / 128, DM / 128);      // 32 x 8 blocks
  gemm128<true,  true ><<<g, 256, 0, stream>>>(xh, Wqe, blq, lqb, DM, DM);
  gemm128<true,  true ><<<g, 256, 0, stream>>>(xh, Wke, blk, lkb, DM, DM);
  gemm128<false, true ><<<g, 256, 0, stream>>>(xh, Wvh, nullptr, vb, DM, DM);

  flash_attn<<<BATCH * NH * (SEQ / 128), 256, 0, stream>>>(lqb, lkb, vb, ab);

  gemm128<false, false><<<g, 256, 0, stream>>>(ab, Woh, nullptr, d_out, DM, DM);
}
